// CreniqQuantizer_84524956385757
// MI455X (gfx1250) — matrix operand order repaired and checked by reading, not compiled, not run on hardware
//
#include <hip/hip_runtime.h>
#include <hip/hip_bf16.h>

// ---------------------------------------------------------------------------
// Problem constants (from reference)
// ---------------------------------------------------------------------------
constexpr int NCB   = 8;      // number of codebooks
constexpr int KC    = 1024;   // codes per codebook
constexpr int SDIM  = 64;     // sub-dim per codebook
constexpr int LDIM  = 512;    // NCB * SDIM
constexpr int BROWS = 8192;   // batch
constexpr float TAU = 0.07f;
constexpr size_t QELEMS = (size_t)BROWS * LDIM;   // 4194304

// d_out layout: [quant_z_st (QELEMS)][commit (1)][contrastive (1)][indices (B*NC)]

typedef __attribute__((ext_vector_type(16))) __bf16 v16bf;
typedef __attribute__((ext_vector_type(8)))  float  v8f;

union Frag16 {
    v16bf v;
    uint4 q[2];
    unsigned short s[16];
};

__device__ __forceinline__ unsigned short f2bf(float f) {
    union { float f; unsigned u; } x; x.f = f;
    unsigned r = x.u + 0x7fffu + ((x.u >> 16) & 1u);   // round-to-nearest-even
    return (unsigned short)(r >> 16);
}

// CDNA5 async DMA: global -> LDS, 16B per lane, tracked by ASYNCcnt.
// First operand: per-lane LDS byte offset (workgroup-relative); second: 64-bit VA.
__device__ __forceinline__ void async_ld16(unsigned lds_off, const void* g) {
    asm volatile("global_load_async_to_lds_b128 %0, %1, off"
                 :: "v"(lds_off), "v"((unsigned long long)(size_t)g)
                 : "memory");
}

// ---------------------------------------------------------------------------
// K0: embedding prep — e2[n,k] = ||emb[n,k]||^2, bf16 copy of embeddings
// ---------------------------------------------------------------------------
__global__ void __launch_bounds__(256) k_embprep(const float* __restrict__ emb,
                                                 float* __restrict__ e2,
                                                 unsigned short* __restrict__ embbf) {
    int t = blockIdx.x * 256 + threadIdx.x;          // 0 .. NCB*KC-1
    const float* row = emb + (size_t)t * SDIM;
    unsigned short* obf = embbf + (size_t)t * SDIM;
    float ss = 0.f;
    #pragma unroll
    for (int d = 0; d < SDIM; ++d) {
        float f = row[d];
        ss += f * f;
        obf[d] = f2bf(f);
    }
    e2[t] = ss;
}

// ---------------------------------------------------------------------------
// K1: z prep — bf16 copy + row-normalized bf16 copy (one wave per row)
// ---------------------------------------------------------------------------
__global__ void __launch_bounds__(256) k_zprep(const float* __restrict__ z,
                                               unsigned short* __restrict__ zbf,
                                               unsigned short* __restrict__ znbf) {
    int wave = threadIdx.x >> 5;
    int lane = threadIdx.x & 31;
    int row  = blockIdx.x * 8 + wave;
    const float* zr = z + (size_t)row * LDIM;
    float v[16];
    float ss = 0.f;
    #pragma unroll
    for (int i = 0; i < 16; ++i) { v[i] = zr[lane + 32 * i]; ss += v[i] * v[i]; }
    #pragma unroll
    for (int m = 1; m < 32; m <<= 1) ss += __shfl_xor(ss, m, 32);
    float inv = 1.0f / fmaxf(sqrtf(ss), 1e-12f);
    size_t base = (size_t)row * LDIM + lane;
    #pragma unroll
    for (int i = 0; i < 16; ++i) {
        zbf[base + 32 * i]  = f2bf(v[i]);
        znbf[base + 32 * i] = f2bf(v[i] * inv);
    }
}

// ---------------------------------------------------------------------------
// K2: argmin over codes (bf16 WMMA), quantize gather, indices, commit partial
//     block = 256 (8 waves); wave w handles codebook n=w for a 16-row strip
// ---------------------------------------------------------------------------
__global__ void __launch_bounds__(256) k_argmin_quant(
        const float* __restrict__ z, const float* __restrict__ emb,
        const unsigned short* __restrict__ zbf,
        const unsigned short* __restrict__ embbf,
        const float* __restrict__ e2,
        float* __restrict__ quant, float* __restrict__ idxout,
        float* __restrict__ acc) {
    const int n    = threadIdx.x >> 5;    // codebook == wave id
    const int lane = threadIdx.x & 31;
    const int r15  = lane & 15;
    const int half = lane >> 4;
    const int m0   = blockIdx.x * 16;

    // A fragments: 16 rows x dims [0..31] and [32..63]   (ISA 16-bit A layout)
    const unsigned short* arow = zbf + (size_t)(m0 + r15) * LDIM + n * SDIM;
    const int ka = half * 8;
    Frag16 a0, a1;
    a0.q[0] = *(const uint4*)(arow + ka);
    a0.q[1] = *(const uint4*)(arow + ka + 16);
    a1.q[0] = *(const uint4*)(arow + 32 + ka);
    a1.q[1] = *(const uint4*)(arow + 32 + ka + 16);

    float best[8];
    int   bidx[8];
    #pragma unroll
    for (int r = 0; r < 8; ++r) { best[r] = 3.4e38f; bidx[r] = 0; }

    const unsigned short* ebase = embbf + (size_t)n * KC * SDIM;
    const float* e2n = e2 + n * KC;

    for (int tile = 0; tile < KC / 16; ++tile) {
        const int code = tile * 16 + r15;            // this lane's column/code
        const unsigned short* brow = ebase + (size_t)code * SDIM + half * 8;
        Frag16 b0, b1;                               // ISA 16-bit B 32x16 layout
        b0.q[0] = *(const uint4*)(brow);
        b0.q[1] = *(const uint4*)(brow + 16);
        b1.q[0] = *(const uint4*)(brow + 32);
        b1.q[1] = *(const uint4*)(brow + 48);
        v8f c = {};
        c = __builtin_amdgcn_wmma_f32_16x16x32_bf16(false, a0.v, false, b0.v,
                                                    (short)0, c, false, false);
        c = __builtin_amdgcn_wmma_f32_16x16x32_bf16(false, a1.v, false, b1.v,
                                                    (short)0, c, false, false);
        const float e2c = e2n[code];
        #pragma unroll
        for (int r = 0; r < 8; ++r) {
            float s = e2c - 2.0f * c[r];             // z2 constant -> dropped
            if (s < best[r]) { best[r] = s; bidx[r] = code; }
        }
    }

    // min-reduce (dist, idx) across the 16-lane column group
    #pragma unroll
    for (int m = 1; m < 16; m <<= 1) {
        #pragma unroll
        for (int r = 0; r < 8; ++r) {
            float ob = __shfl_xor(best[r], m, 32);
            int   oi = __shfl_xor(bidx[r], m, 32);
            if (ob < best[r] || (ob == best[r] && oi < bidx[r])) {
                best[r] = ob; bidx[r] = oi;
            }
        }
    }

    // indices output (float-cast); lanes 0 / 16 hold rows 0..7 / 8..15
    if (r15 == 0) {
        #pragma unroll
        for (int r = 0; r < 8; ++r)
            idxout[(size_t)(m0 + half * 8 + r) * NCB + n] = (float)bidx[r];
    }

    // quantize gather + commit partial; each lane copies 2 dims per row
    float sq = 0.f;
    #pragma unroll
    for (int rr = 0; rr < 16; ++rr) {
        int code = __shfl(bidx[rr & 7], (rr & 8) ? 16 : 0, 32);
        const float* src = emb + ((size_t)n * KC + code) * SDIM + lane * 2;
        float2 qv = *(const float2*)src;
        size_t o = (size_t)(m0 + rr) * LDIM + n * SDIM + lane * 2;
        float2 zv = *(const float2*)(z + o);
        *(float2*)(quant + o) = qv;
        float dx = qv.x - zv.x, dy = qv.y - zv.y;
        sq += dx * dx + dy * dy;
    }
    #pragma unroll
    for (int m = 1; m < 32; m <<= 1) sq += __shfl_xor(sq, m, 32);
    if (lane == 0) atomicAdd(&acc[0], sq);
}

// ---------------------------------------------------------------------------
// K3: quantized-row normalization -> qn bf16 (one wave per row)
// ---------------------------------------------------------------------------
__global__ void __launch_bounds__(256) k_qprep(const float* __restrict__ q,
                                               unsigned short* __restrict__ qnbf) {
    int wave = threadIdx.x >> 5;
    int lane = threadIdx.x & 31;
    int row  = blockIdx.x * 8 + wave;
    const float* qr = q + (size_t)row * LDIM;
    float v[16];
    float ss = 0.f;
    #pragma unroll
    for (int i = 0; i < 16; ++i) { v[i] = qr[lane + 32 * i]; ss += v[i] * v[i]; }
    #pragma unroll
    for (int m = 1; m < 32; m <<= 1) ss += __shfl_xor(ss, m, 32);
    float inv = 1.0f / fmaxf(sqrtf(ss), 1e-12f);
    size_t base = (size_t)row * LDIM + lane;
    #pragma unroll
    for (int i = 0; i < 16; ++i) qnbf[base + 32 * i] = f2bf(v[i] * inv);
}

// ---------------------------------------------------------------------------
// K4: fused contrastive — flash-style logsumexp over 8192x8192 logits.
//     block = 256 (8 waves) covers 128 rows. A rows staged in LDS once;
//     64-col B tiles double-buffered via GLOBAL_LOAD_ASYNC_TO_LDS (ASYNCcnt).
//     LDS: A 128x520 bf16 (133,120 B) + 2 x (64x520 bf16) = 266,240 B.
// ---------------------------------------------------------------------------
constexpr int CT      = 64;            // column tile
constexpr int NT      = BROWS / CT;    // 128 tiles
constexpr int LDSTR   = 520;           // padded row stride (elems) — bank spread
constexpr int A_ELEMS = 128 * LDSTR;   // 66560 elems
constexpr unsigned ABYTES = A_ELEMS * 2;        // 133120
constexpr unsigned BBYTES = CT * LDSTR * 2;     // 66560
constexpr size_t LDS_TOTAL = ABYTES + 2 * (size_t)BBYTES;   // 266240

__device__ __forceinline__ void issue_btile(const unsigned short* zn, int tile,
                                            unsigned lbytes, int tid) {
    // 64 rows x 512 bf16 = 4096 x 16B ; 256 threads -> 16 async loads each
    #pragma unroll
    for (int i = 0; i < 16; ++i) {
        int u   = tid + 256 * i;
        int row = u >> 6;            // 64 uint4 per row
        int c8  = u & 63;
        unsigned loff = lbytes + (unsigned)(row * LDSTR + c8 * 8) * 2;
        async_ld16(loff, zn + ((size_t)(tile * CT + row)) * LDIM + c8 * 8);
    }
}

__global__ void __launch_bounds__(256) k_contrast(
        const unsigned short* __restrict__ qn,
        const unsigned short* __restrict__ zn,
        float* __restrict__ acc) {
    extern __shared__ unsigned short lds[];
    const int tid  = threadIdx.x;
    const int lane = tid & 31;
    const int wave = tid >> 5;
    const int r15  = lane & 15;
    const int half = lane >> 4;
    const int rowbase = blockIdx.x * 128 + wave * 16;
    const float invTau = 1.0f / TAU;
    const unsigned lbase = (unsigned)(size_t)(void*)lds;  // LDS byte offset

    // stage A block (128 rows x 512 bf16) + first B tile via async DMA
    #pragma unroll
    for (int i = 0; i < 32; ++i) {
        int u   = tid + 256 * i;
        int row = u >> 6;
        int c8  = u & 63;
        unsigned loff = lbase + (unsigned)(row * LDSTR + c8 * 8) * 2;
        async_ld16(loff, qn + ((size_t)(blockIdx.x * 128 + row)) * LDIM + c8 * 8);
    }
    issue_btile(zn, 0, lbase + ABYTES, tid);

    float m[8], s[8], dg[8];
    #pragma unroll
    for (int r = 0; r < 8; ++r) { m[r] = -3.4e38f; s[r] = 0.f; dg[r] = 0.f; }

    const unsigned short* aL = lds + (wave * 16 + r15) * LDSTR;

    for (int ct = 0; ct < NT; ++ct) {
        __syncthreads();    // previous iteration's readers of the next buffer done
        if (ct + 1 < NT) {
            issue_btile(zn, ct + 1,
                        lbase + ABYTES + (unsigned)((ct + 1) & 1) * BBYTES, tid);
            // next tile (16) may stay in flight; current tile guaranteed resident
            asm volatile("s_wait_asynccnt 0x10" ::: "memory");
        } else {
            asm volatile("s_wait_asynccnt 0x0" ::: "memory");
        }
        __syncthreads();    // async writes visible to all waves

        const unsigned short* bL = lds + A_ELEMS + (size_t)(ct & 1) * (CT * LDSTR);

        v8f c[4];
        #pragma unroll
        for (int j = 0; j < 4; ++j) c[j] = (v8f){};

        for (int kk = 0; kk < LDIM; kk += 32) {
            Frag16 a;
            const unsigned short* ap = aL + kk + half * 8;
            a.q[0] = *(const uint4*)(ap);
            a.q[1] = *(const uint4*)(ap + 16);
            #pragma unroll
            for (int j = 0; j < 4; ++j) {
                Frag16 b;
                const unsigned short* bp = bL + (j * 16 + r15) * LDSTR + kk + half * 8;
                b.q[0] = *(const uint4*)(bp);
                b.q[1] = *(const uint4*)(bp + 16);
                c[j] = __builtin_amdgcn_wmma_f32_16x16x32_bf16(false, a.v, false, b.v,
                                                               (short)0, c[j], false, false);
            }
        }

        // online logsumexp update per owned row
        #pragma unroll
        for (int r = 0; r < 8; ++r) {
            const int grow = rowbase + half * 8 + r;
            float tm = -3.4e38f;
            #pragma unroll
            for (int j = 0; j < 4; ++j) tm = fmaxf(tm, c[j][r]);
            tm *= invTau;
            #pragma unroll
            for (int mm = 1; mm < 16; mm <<= 1) tm = fmaxf(tm, __shfl_xor(tm, mm, 32));
            float mn = fmaxf(m[r], tm);
            float p = 0.f;
            #pragma unroll
            for (int j = 0; j < 4; ++j) {
                float lv = c[j][r] * invTau;
                p += __expf(lv - mn);
                int gcol = ct * CT + j * 16 + r15;
                if (gcol == grow) dg[r] = lv;        // diagonal capture
            }
            #pragma unroll
            for (int mm = 1; mm < 16; mm <<= 1) p += __shfl_xor(p, mm, 32);
            s[r] = s[r] * __expf(m[r] - mn) + p;
            m[r] = mn;
        }
    }

    float tot = 0.f;
    #pragma unroll
    for (int r = 0; r < 8; ++r) {
        float d = dg[r];
        #pragma unroll
        for (int mm = 1; mm < 16; mm <<= 1) d += __shfl_xor(d, mm, 32); // one lane set it
        tot += (m[r] + __logf(s[r])) - d;
    }
    if (r15 == 0) atomicAdd(&acc[1], tot);
}

// ---------------------------------------------------------------------------
// K5: finalize scalars
// ---------------------------------------------------------------------------
__global__ void k_final(const float* __restrict__ acc, float* __restrict__ out) {
    if (threadIdx.x == 0) {
        out[QELEMS]     = 2.0f * acc[0] / (float)(BROWS * LDIM);  // commit+embed
        out[QELEMS + 1] = acc[1] / (float)BROWS;                  // contrastive
    }
}

// ---------------------------------------------------------------------------
// Launch
// ---------------------------------------------------------------------------
extern "C" void kernel_launch(void* const* d_in, const int* in_sizes, int n_in,
                              void* d_out, int out_size, void* d_ws, size_t ws_size,
                              hipStream_t stream) {
    const float* z   = (const float*)d_in[0];
    const float* emb = (const float*)d_in[1];
    float* out   = (float*)d_out;
    float* quant = out;                       // QELEMS
    float* idxo  = out + QELEMS + 2;          // B*NC indices (as float)

    char* ws = (char*)d_ws;
    float*          acc   = (float*)ws;                          // 2 floats (+pad)
    float*          e2    = (float*)(ws + 256);                  // NC*KC f32
    unsigned short* embbf = (unsigned short*)(ws + 256 + 32768); // NC*KC*SD bf16
    unsigned short* zbf   = (unsigned short*)(ws + 1081600);     // B*L bf16
    unsigned short* znbf  = (unsigned short*)(ws + 9470208);     // B*L bf16
    unsigned short* qnbf  = (unsigned short*)(ws + 17858816);    // B*L bf16

    hipMemsetAsync(acc, 0, 256, stream);

    k_embprep<<<(NCB * KC) / 256, 256, 0, stream>>>(emb, e2, embbf);
    k_zprep<<<BROWS / 8, 256, 0, stream>>>(z, zbf, znbf);
    k_argmin_quant<<<BROWS / 16, 256, 0, stream>>>(z, emb, zbf, embbf, e2,
                                                   quant, idxo, acc);
    k_qprep<<<BROWS / 8, 256, 0, stream>>>(quant, qnbf);
    k_contrast<<<BROWS / 128, 256, LDS_TOTAL, stream>>>(qnbf, znbf, acc);
    k_final<<<1, 32, 0, stream>>>(acc, out);
}
